// DecoderBlock_64896955842946
// MI455X (gfx1250) — matrix operand order repaired and checked by reading, not compiled, not run on hardware
//
#include <hip/hip_runtime.h>
#include <cstdint>
#include <cstddef>

typedef __attribute__((ext_vector_type(16))) __bf16 v16bf;
typedef __attribute__((ext_vector_type(8)))  __bf16 v8bf;
typedef __attribute__((ext_vector_type(8)))  float  v8f;

#define DEV __device__ __forceinline__

#if defined(__HIP_DEVICE_COMPILE__) && __has_builtin(__builtin_amdgcn_tensor_load_to_lds) && __has_builtin(__builtin_amdgcn_s_wait_tensorcnt)
#define HAVE_TDM 1
#else
#define HAVE_TDM 0
#endif

DEV __bf16 f2bf(float f) {
    unsigned u = __float_as_uint(f);
    u += 0x7FFFu + ((u >> 16) & 1u);          // round-to-nearest-even
    unsigned short s = (unsigned short)(u >> 16);
    union { unsigned short us; __bf16 b; } cv; cv.us = s; return cv.b;
}

DEV v16bf cat8(v8bf lo, v8bf hi) {
    return __builtin_shufflevector(lo, hi, 0,1,2,3,4,5,6,7,8,9,10,11,12,13,14,15);
}
// A-frag: lane m=lane%16, half=lane/16; elems h<8 at [half*8+h], h>=8 at [16+half*8+(h-8)]
DEV v16bf ldfragA(const __bf16* p) {          // p = &tile[m*stride + half*8]
    v8bf l0 = *(const v8bf*)p;
    v8bf l1 = *(const v8bf*)(p + 16);
    return cat8(l0, l1);
}
// B-frag: lane n=lane%16, half=lane/16; elems h<8 at [half*8+h], h>=8 at [16+half*8+(h-8)]
DEV v16bf ldfragB(const __bf16* p) {          // p = &tileT[n*stride + half*8]
    v8bf l0 = *(const v8bf*)p;
    v8bf l1 = *(const v8bf*)(p + 16);
    return cat8(l0, l1);
}
DEV v8f wmma_bf16(v16bf a, v16bf b, v8f c) {
    return __builtin_amdgcn_wmma_f32_16x16x32_bf16(false, a, false, b, (short)0, c, false, false);
}

// ---- 16-lane butterfly reductions in pure VALU (DPP), no LDS traffic ----
#define DPP_STEP(v, op, ctrl)                                                   \
    {                                                                           \
        int _x = __float_as_int(v);                                             \
        float _t = __int_as_float(                                              \
            __builtin_amdgcn_update_dpp(_x, _x, ctrl, 0xF, 0xF, true));         \
        v = op(v, _t);                                                          \
    }
DEV float rmax16(float v) {
    DPP_STEP(v, fmaxf, 0xB1);   // quad_perm(1,0,3,2)  : xor 1
    DPP_STEP(v, fmaxf, 0x4E);   // quad_perm(2,3,0,1)  : xor 2
    DPP_STEP(v, fmaxf, 0x124);  // row_ror:4 (quads uniform -> pairs quads)
    DPP_STEP(v, fmaxf, 0x128);  // row_ror:8
    return v;
}
DEV float addf(float a, float b) { return a + b; }
DEV float rsum16(float v) {
    DPP_STEP(v, addf, 0xB1);
    DPP_STEP(v, addf, 0x4E);
    DPP_STEP(v, addf, 0x124);
    DPP_STEP(v, addf, 0x128);
    return v;
}

#if HAVE_TDM
typedef __attribute__((ext_vector_type(4))) unsigned tdmU4;
typedef __attribute__((ext_vector_type(8))) int      tdmI8;
typedef __attribute__((ext_vector_type(4))) int      tdmI4;
// DMA one [128 rows x 32 bf16] tile (row stride = stride_elems) into LDS at
// lds_off, LDS row pitch 40 bf16 (16 DWORDs data + 4 DWORDs pad via D# padding).
DEV void tdm_load_tile(const __bf16* gsrc, unsigned lds_off, unsigned stride_elems) {
    unsigned long long ga = (unsigned long long)(size_t)gsrc;
    unsigned td0 = stride_elems;
    tdmU4 g0;
    g0[0] = 1u;                                   // count=1 (valid), user mode
    g0[1] = lds_off;                              // lds_addr
    g0[2] = (unsigned)ga;                         // global_addr[31:0]
    g0[3] = (unsigned)((ga >> 32) & 0x01FFFFFFu)  // global_addr[56:32]
            | (2u << 30);                         // type=2 ("image")
    tdmI8 g1;
    g1[0] = (int)((1u << 16)      // data_size = 1 (2 bytes)
                | (1u << 20)      // pad_enable
                | (3u << 22)      // pad_interval = 3 -> every 16 DWORDs
                | (3u << 25));    // pad_amount   = 3 -> 4 DWORDs
    g1[1] = (int)(td0 << 16);                     // tensor_dim0[15:0]
    g1[2] = (int)((td0 >> 16) & 0xFFFFu);         // tensor_dim0[31:16], dim1 lo=0
    g1[3] = (int)(16u | (32u << 16));             // tensor_dim1 = 1<<20, tile_dim0 = 32
    g1[4] = (int)128;                             // tile_dim1 = 128, tile_dim2 = 0
    g1[5] = (int)td0;                             // tensor_dim0_stride[31:0]
    g1[6] = 0;                                    // stride hi / dim1_stride lo
    g1[7] = 0;
    tdmI4 z4; z4[0] = 0; z4[1] = 0; z4[2] = 0; z4[3] = 0;
    tdmI8 z8;
#pragma unroll
    for (int i = 0; i < 8; i++) z8[i] = 0;
    // 6-arg toolchain form: (g0, g1, g2, g3, g4, cpol)
    __builtin_amdgcn_tensor_load_to_lds(g0, g1, z4, z4, z8, 0);
}
DEV unsigned lds_off32(const void* p) {           // flat LDS addr: low 32 bits
    return (unsigned)(size_t)p;
}
#endif

// ---------------- fp32 -> bf16 transposed weight conversion ----------------
// src: K x Nn row-major, dst: Nn x K row-major (so GEMM B tiles are row copies)
__global__ __launch_bounds__(256) void cvt_t_kernel(const float* __restrict__ s,
                                                    __bf16* __restrict__ d,
                                                    int K, int Nn) {
    int n = blockIdx.x * 16 + (threadIdx.x & 15);
    int k = blockIdx.y * 16 + (threadIdx.x >> 4);
    d[(size_t)n * K + k] = f2bf(s[(size_t)k * Nn + n]);
}

// ---------------- LayerNorm (C = 768), wave per row ----------------
__global__ __launch_bounds__(256) void ln_kernel(const float* __restrict__ x,
                                                 const float* __restrict__ g,
                                                 const float* __restrict__ be,
                                                 __bf16* __restrict__ out) {
    const int C = 768;
    int wid = threadIdx.x >> 5, lane = threadIdx.x & 31;
    int row = blockIdx.x * 8 + wid;
    const float* xr = x + (size_t)row * C;
    float v[24], s = 0.f;
#pragma unroll
    for (int i = 0; i < 24; i++) { v[i] = xr[lane + 32 * i]; s += v[i]; }
#pragma unroll
    for (int m = 16; m >= 1; m >>= 1) s += __shfl_xor(s, m, 32);
    float mean = s * (1.f / 768.f), vs = 0.f;
#pragma unroll
    for (int i = 0; i < 24; i++) { float d = v[i] - mean; vs += d * d; }
#pragma unroll
    for (int m = 16; m >= 1; m >>= 1) vs += __shfl_xor(vs, m, 32);
    float rstd = rsqrtf(vs * (1.f / 768.f) + 1e-5f);
    __bf16* o = out + (size_t)row * C;
#pragma unroll
    for (int i = 0; i < 24; i++) {
        int c = lane + 32 * i;
        o[c] = f2bf((v[i] - mean) * rstd * g[c] + be[c]);
    }
}

// ---------------- block GEMM: 128x128 tile, K-step 32, bf16 WMMA ----------------
// Both A and B tiles are [128 rows x 32 elems], row pitch 40 in LDS.
DEV void stage_tile(const __bf16* __restrict__ src, int ld, __bf16* dst, int tid) {
    int row = tid >> 1, ch = (tid & 1) * 16;
    const __bf16* s = src + (size_t)row * ld + ch;
    v8bf lo = *(const v8bf*)s, hi = *(const v8bf*)(s + 8);
    *(v8bf*)(dst + row * 40 + ch) = lo;
    *(v8bf*)(dst + row * 40 + ch + 8) = hi;
}

template <bool BIAS, bool RES, bool GELU, bool OUTF32>
__global__ __launch_bounds__(256) void gemm_bf16(
    const __bf16* __restrict__ A, const __bf16* __restrict__ Bt,
    const float* __restrict__ bias, const float* __restrict__ res,
    void* __restrict__ Out, int Nn, int K) {
    __shared__ __bf16 Asm[2][128 * 40];
    __shared__ __bf16 Bsm[2][128 * 40];
    const int lda = K, ldb = K, ldc = Nn;
    int tid = threadIdx.x, wid = tid >> 5, lane = tid & 31;
    int bm = blockIdx.y * 128, bn = blockIdx.x * 128;
    int wr = wid >> 1, wc = wid & 1;
    int mlane = lane & 15, half = lane >> 4;
    const __bf16* Abase = A + (size_t)bm * lda;
    const __bf16* Bbase = Bt + (size_t)bn * ldb;

    v8f acc[2][4];
#pragma unroll
    for (int i = 0; i < 2; i++)
#pragma unroll
        for (int j = 0; j < 4; j++) { v8f z = {}; acc[i][j] = z; }

#if HAVE_TDM
    if (wid == 0) {
        tdm_load_tile(Abase, lds_off32(&Asm[0][0]), lda);
        tdm_load_tile(Bbase, lds_off32(&Bsm[0][0]), ldb);
    }
#else
    stage_tile(Abase, lda, Asm[0], tid);
    stage_tile(Bbase, ldb, Bsm[0], tid);
#endif
    int nk = K / 32;
    for (int kk = 0; kk < nk; kk++) {
        int buf = kk & 1;
#if HAVE_TDM
        if (wid == 0) __builtin_amdgcn_s_wait_tensorcnt(0);
#endif
        __syncthreads();
        if (kk + 1 < nk) {
            int k1 = (kk + 1) * 32;
#if HAVE_TDM
            if (wid == 0) {
                tdm_load_tile(Abase + k1, lds_off32(&Asm[buf ^ 1][0]), lda);
                tdm_load_tile(Bbase + k1, lds_off32(&Bsm[buf ^ 1][0]), ldb);
            }
#else
            stage_tile(Abase + k1, lda, Asm[buf ^ 1], tid);
            stage_tile(Bbase + k1, ldb, Bsm[buf ^ 1], tid);
#endif
        }
        v16bf a[2], b[4];
#pragma unroll
        for (int i = 0; i < 2; i++)
            a[i] = ldfragA(&Asm[buf][(wr * 32 + i * 16 + mlane) * 40 + half * 8]);
#pragma unroll
        for (int j = 0; j < 4; j++)
            b[j] = ldfragB(&Bsm[buf][(wc * 64 + j * 16 + mlane) * 40 + half * 8]);
#pragma unroll
        for (int i = 0; i < 2; i++)
#pragma unroll
            for (int j = 0; j < 4; j++)
                acc[i][j] = wmma_bf16(a[i], b[j], acc[i][j]);
    }

#pragma unroll
    for (int i = 0; i < 2; i++)
#pragma unroll
        for (int j = 0; j < 4; j++) {
            int col = bn + wc * 64 + j * 16 + mlane;
            float bv = BIAS ? bias[col] : 0.f;
#pragma unroll
            for (int r = 0; r < 8; r++) {
                int row = bm + wr * 32 + i * 16 + half * 8 + r;
                float v = acc[i][j][r] + bv;
                if (GELU) v = 0.5f * v * (1.f + erff(v * 0.70710678118654752f));
                if (RES)  v += res[(size_t)row * ldc + col];
                if (OUTF32) ((float*)Out)[(size_t)row * ldc + col] = v;
                else        ((__bf16*)Out)[(size_t)row * ldc + col] = f2bf(v);
            }
        }
}

// ---------------- flash attention: D=64, H=12, N=2048 ----------------
__global__ __launch_bounds__(256) void attn_kernel(
    const __bf16* __restrict__ Q, int sq,
    const __bf16* __restrict__ Kp, int sk,
    const __bf16* __restrict__ Vp, int sv,
    __bf16* __restrict__ O, float scale) {
    const int N = 2048, H = 12;
    __shared__ __bf16 Ks[32 * 72];       // 32 keys x 64 d (stride 72)
    __shared__ __bf16 Vt[64 * 40];       // transposed: 64 d x 32 k (stride 40)
    __shared__ __bf16 Ps[8][16 * 40];    // per-wave P tile 16x32 (stride 40)

    int tid = threadIdx.x, wid = tid >> 5, lane = tid & 31;
    int bh = blockIdx.y, b = bh / H, hh = bh % H;
    int q0 = blockIdx.x * 128 + wid * 16;
    int mlane = lane & 15, half = lane >> 4;

    // Q fragments straight from global in A-layout
    const __bf16* qbase = Q + (size_t)(b * N + q0 + mlane) * sq + hh * 64;
    v16bf qf[2];
#pragma unroll
    for (int ds = 0; ds < 2; ds++) {
        v8bf l0 = *(const v8bf*)(qbase + ds * 32 + half * 8);
        v8bf l1 = *(const v8bf*)(qbase + ds * 32 + 16 + half * 8);
        qf[ds] = cat8(l0, l1);
    }

    v8f o[4];
#pragma unroll
    for (int j = 0; j < 4; j++) { v8f z = {}; o[j] = z; }
    float mprev[8], lsum[8];
#pragma unroll
    for (int r = 0; r < 8; r++) { mprev[r] = -INFINITY; lsum[r] = 0.f; }

    for (int j0 = 0; j0 < N; j0 += 32) {
        __syncthreads();
        {   // stage K row-major
            int r = tid >> 3, dOff = (tid & 7) * 8;
            const __bf16* src = Kp + (size_t)(b * N + j0 + r) * sk + hh * 64 + dOff;
            *(v8bf*)(Ks + r * 72 + dOff) = *(const v8bf*)src;
        }
        {   // stage V transposed
            int r = tid >> 3, dOff = (tid & 7) * 8;
            const __bf16* src = Vp + (size_t)(b * N + j0 + r) * sv + hh * 64 + dOff;
            v8bf v = *(const v8bf*)src;
#pragma unroll
            for (int t = 0; t < 8; t++) Vt[(dOff + t) * 40 + r] = v[t];
        }
        __syncthreads();

        // S = Q K^T  (two 16-col subtiles)
        v8f s[2];
        { v8f z = {}; s[0] = z; s[1] = z; }
#pragma unroll
        for (int ct = 0; ct < 2; ct++)
#pragma unroll
            for (int ds = 0; ds < 2; ds++) {
                v16bf kb = ldfragB(&Ks[(ct * 16 + mlane) * 72 + ds * 32 + half * 8]);
                s[ct] = wmma_bf16(qf[ds], kb, s[ct]);
            }

        // online softmax (row M = half*8 + r lives in one 16-lane half)
        float alpha[8];
#pragma unroll
        for (int r = 0; r < 8; r++) {
            float a0 = s[0][r] * scale, a1 = s[1][r] * scale;
            s[0][r] = a0; s[1][r] = a1;
            float mx = rmax16(fmaxf(a0, a1));
            float mn = fmaxf(mprev[r], mx);
            alpha[r] = __expf(mprev[r] - mn);
            mprev[r] = mn;
        }
#pragma unroll
        for (int r = 0; r < 8; r++) {
            float p0 = __expf(s[0][r] - mprev[r]);
            float p1 = __expf(s[1][r] - mprev[r]);
            lsum[r] = lsum[r] * alpha[r] + rsum16(p0 + p1);
            int m = half * 8 + r;
            Ps[wid][m * 40 + mlane]      = f2bf(p0);
            Ps[wid][m * 40 + 16 + mlane] = f2bf(p1);
            o[0][r] *= alpha[r]; o[1][r] *= alpha[r];
            o[2][r] *= alpha[r]; o[3][r] *= alpha[r];
        }
        // P back as an A-fragment (same-wave LDS, DS ops are in-order)
        v16bf pf = ldfragA(&Ps[wid][mlane * 40 + half * 8]);
#pragma unroll
        for (int jt = 0; jt < 4; jt++) {
            v16bf vb = ldfragB(&Vt[(jt * 16 + mlane) * 40 + half * 8]);
            o[jt] = wmma_bf16(pf, vb, o[jt]);
        }
    }

    __bf16* ob = O + (size_t)(b * N + q0) * 768 + hh * 64;
#pragma unroll
    for (int r = 0; r < 8; r++) {
        float inv = 1.f / lsum[r];
        int m = half * 8 + r;
#pragma unroll
        for (int jt = 0; jt < 4; jt++)
            ob[(size_t)m * 768 + jt * 16 + mlane] = f2bf(o[jt][r] * inv);
    }
}

// ---------------- host orchestration ----------------
extern "C" void kernel_launch(void* const* d_in, const int* in_sizes, int n_in,
                              void* d_out, int out_size, void* d_ws, size_t ws_size,
                              hipStream_t stream) {
    (void)in_sizes; (void)n_in; (void)out_size; (void)ws_size;
    const int Bb = 4, N = 2048, C = 768, HID = 3072;
    const int M = Bb * N;                 // 8192
    const float* x    = (const float*)d_in[0];
    const float* y    = (const float*)d_in[1];
    const float* g1   = (const float*)d_in[4];
    const float* be1  = (const float*)d_in[5];
    const float* g2   = (const float*)d_in[6];
    const float* be2  = (const float*)d_in[7];
    const float* g3   = (const float*)d_in[8];
    const float* be3  = (const float*)d_in[9];
    const float* gy   = (const float*)d_in[10];
    const float* bey  = (const float*)d_in[11];
    const float* Wqkv = (const float*)d_in[12];
    const float* Wosa = (const float*)d_in[13];
    const float* bosa = (const float*)d_in[14];
    const float* Wq   = (const float*)d_in[15];
    const float* Wk   = (const float*)d_in[16];
    const float* Wv   = (const float*)d_in[17];
    const float* Woca = (const float*)d_in[18];
    const float* boca = (const float*)d_in[19];
    const float* W1   = (const float*)d_in[20];
    const float* b1   = (const float*)d_in[21];
    const float* W2   = (const float*)d_in[22];
    const float* b2   = (const float*)d_in[23];

    char* p = (char*)d_ws;
    auto carve = [&](size_t bytes) -> char* {
        char* r = p; p += (bytes + 255) & ~(size_t)255; return r;
    };
    __bf16* wqkv_bf = (__bf16*)carve((size_t)C * 3 * C * 2);   // transposed: 3C x C
    __bf16* wosa_bf = (__bf16*)carve((size_t)C * C * 2);
    __bf16* wq_bf   = (__bf16*)carve((size_t)C * C * 2);
    __bf16* wk_bf   = (__bf16*)carve((size_t)C * C * 2);
    __bf16* wv_bf   = (__bf16*)carve((size_t)C * C * 2);
    __bf16* woca_bf = (__bf16*)carve((size_t)C * C * 2);
    __bf16* w1_bf   = (__bf16*)carve((size_t)C * HID * 2);     // transposed: HID x C
    __bf16* w2_bf   = (__bf16*)carve((size_t)HID * C * 2);     // transposed: C x HID
    __bf16* ln_bf   = (__bf16*)carve((size_t)M * C * 2);       // xn / xn2 / xn3
    __bf16* yn_bf   = (__bf16*)carve((size_t)M * C * 2);
    __bf16* qkv_bf  = (__bf16*)carve((size_t)M * 3 * C * 2);
    __bf16* qc_bf   = (__bf16*)carve((size_t)M * C * 2);
    __bf16* kc_bf   = (__bf16*)carve((size_t)M * C * 2);
    __bf16* vc_bf   = (__bf16*)carve((size_t)M * C * 2);
    __bf16* attn_bf = (__bf16*)carve((size_t)M * C * 2);
    float*  x2_f    = (float*)carve((size_t)M * C * 4);
    float*  x3_f    = (float*)carve((size_t)M * C * 4);
    __bf16* h_bf    = (__bf16*)carve((size_t)M * HID * 2);

    auto cvtT = [&](const float* s, __bf16* d, int K, int Nn) {
        cvt_t_kernel<<<dim3(Nn / 16, K / 16), 256, 0, stream>>>(s, d, K, Nn);
    };
    cvtT(Wqkv, wqkv_bf, C, 3 * C);
    cvtT(Wosa, wosa_bf, C, C);
    cvtT(Wq,   wq_bf,   C, C);
    cvtT(Wk,   wk_bf,   C, C);
    cvtT(Wv,   wv_bf,   C, C);
    cvtT(Woca, woca_bf, C, C);
    cvtT(W1,   w1_bf,   C, HID);
    cvtT(W2,   w2_bf,   HID, C);

    const float scale = 0.125f;           // 64^-0.5
    dim3 blk(256);
    dim3 gAttn(N / 128, Bb * 12);

    // ---- self attention ----
    ln_kernel<<<dim3(M / 8), blk, 0, stream>>>(x, g1, be1, ln_bf);
    gemm_bf16<false,false,false,false><<<dim3(3 * C / 128, M / 128), blk, 0, stream>>>(
        ln_bf, wqkv_bf, nullptr, nullptr, qkv_bf, 3 * C, C);
    attn_kernel<<<gAttn, blk, 0, stream>>>(qkv_bf, 3 * C, qkv_bf + C, 3 * C,
                                           qkv_bf + 2 * C, 3 * C, attn_bf, scale);
    gemm_bf16<true,true,false,true><<<dim3(C / 128, M / 128), blk, 0, stream>>>(
        attn_bf, wosa_bf, bosa, x, x2_f, C, C);

    // ---- cross attention ----
    ln_kernel<<<dim3(M / 8), blk, 0, stream>>>(y, gy, bey, yn_bf);
    ln_kernel<<<dim3(M / 8), blk, 0, stream>>>(x2_f, g2, be2, ln_bf);
    gemm_bf16<false,false,false,false><<<dim3(C / 128, M / 128), blk, 0, stream>>>(
        ln_bf, wq_bf, nullptr, nullptr, qc_bf, C, C);
    gemm_bf16<false,false,false,false><<<dim3(C / 128, M / 128), blk, 0, stream>>>(
        yn_bf, wk_bf, nullptr, nullptr, kc_bf, C, C);
    gemm_bf16<false,false,false,false><<<dim3(C / 128, M / 128), blk, 0, stream>>>(
        yn_bf, wv_bf, nullptr, nullptr, vc_bf, C, C);
    attn_kernel<<<gAttn, blk, 0, stream>>>(qc_bf, C, kc_bf, C, vc_bf, C, attn_bf, scale);
    gemm_bf16<true,true,false,true><<<dim3(C / 128, M / 128), blk, 0, stream>>>(
        attn_bf, woca_bf, boca, x2_f, x3_f, C, C);

    // ---- MLP ----
    ln_kernel<<<dim3(M / 8), blk, 0, stream>>>(x3_f, g3, be3, ln_bf);
    gemm_bf16<true,false,true,false><<<dim3(HID / 128, M / 128), blk, 0, stream>>>(
        ln_bf, w1_bf, b1, nullptr, h_bf, HID, C);
    gemm_bf16<true,true,false,true><<<dim3(C / 128, M / 128), blk, 0, stream>>>(
        h_bf, w2_bf, b2, x3_f, (float*)d_out, C, HID);
}
